// Transformer_29927332119124
// MI455X (gfx1250) — matrix operand order repaired and checked by reading, not compiled, not run on hardware
//
#include <hip/hip_runtime.h>
#include <math.h>
#include <stdint.h>

#define DEV __device__ __forceinline__

typedef __bf16 bf16;
typedef __attribute__((ext_vector_type(16))) __bf16 v16bf;
typedef __attribute__((ext_vector_type(8)))  __bf16 v8bf;
typedef __attribute__((ext_vector_type(8)))  float  v8f;
typedef __attribute__((ext_vector_type(4)))  unsigned int u32x4;
typedef __attribute__((ext_vector_type(8)))  unsigned int u32x8;

// ---------------- problem constants (fixed by the reference) ----------------
static constexpr int Bc  = 2;
static constexpr int Tc  = 2048;
static constexpr int Dc  = 1024;
static constexpr int Hc  = 16;
static constexpr int NKc = 64;
static constexpr int NVc = 64;
static constexpr int NFc = 4096;
static constexpr int BTc = Bc * Tc;   // 4096 rows
static constexpr int BHc = Bc * Hc;   // 32 (b,h) pairs
static constexpr float EPSc = 1e-5f;

// ---------------- WMMA helpers (gfx1250, wave32) ----------------
DEV v8f wmma_bf16(v16bf a, v16bf b, v8f c) {
  // D = A(16x32 bf16) x B(32x16 bf16) + C(16x16 f32)
  return __builtin_amdgcn_wmma_f32_16x16x32_bf16(false, a, false, b, (short)0, c,
                                                 false, false);
}

// A-fragment: lane L holds row M = L%16; element e -> K = (e<8 ? e : e+8) + (L>=16 ? 8:0)
DEV v16bf load_a_frag(const bf16* __restrict__ A, size_t lda, size_t row,
                      size_t kbase, int lane) {
  const int hi8 = (lane & 16) ? 8 : 0;
  const bf16* p = A + row * lda + kbase + hi8;
  v8bf lo = *(const v8bf*)(p);        // K = kbase+hi8    .. +7
  v8bf hi = *(const v8bf*)(p + 16);   // K = kbase+16+hi8 .. +7
  v16bf r;
#pragma unroll
  for (int i = 0; i < 8; ++i) { r[i] = lo[i]; r[8 + i] = hi[i]; }
  return r;
}

// B-fragment: lane L holds column N = L%16; element e -> K = (e<8 ? e : e+8) + (L>=16 ? 8:0)
DEV v16bf load_b_frag(const bf16* __restrict__ Bm, size_t ldb, size_t kbase,
                      size_t col, int lane) {
  const int hi8 = (lane & 16) ? 8 : 0;
  const bf16* p = Bm + (kbase + hi8) * ldb + col + (lane & 15);
  v16bf r;
#pragma unroll
  for (int i = 0; i < 8; ++i) { r[i] = p[i * ldb]; r[8 + i] = p[(16 + i) * ldb]; }
  return r;
}

// ---------------- Tensor Data Mover: 2D tile (global -> LDS) ----------------
// Builds a D# per CDNA5 ISA and issues TENSOR_LOAD_TO_LDS
// (2-group form, VADDR2/VADDR3 = NULL -> up to 2D). data_size = 2 bytes (bf16).
// Tile: tile_d1 rows x tile_d0 elements, row stride = stride0 elements;
// packed contiguously into LDS at lds_addr.
DEV void tdm_load_2d_bf16(uint32_t lds_addr, const bf16* gptr,
                          uint32_t tensor_d0, uint32_t tensor_d1,
                          uint32_t tile_d0, uint32_t tile_d1,
                          uint32_t stride0) {
  const uint64_t ga = (uint64_t)(uintptr_t)gptr;
  u32x4 g0;
  g0[0] = 1u;                                            // count=1, user mode
  g0[1] = lds_addr;                                      // lds_addr[31:0]
  g0[2] = (uint32_t)(ga & 0xffffffffu);                  // global_addr[31:0]
  g0[3] = (uint32_t)((ga >> 32) & 0x01ffffffu)           // global_addr[56:32]
          | (2u << 30);                                  // type = 2 ("image")
  u32x8 g1;
  g1[0] = 0x10000u;                                      // data_size=1 (2B), mask=0
  g1[1] = (tensor_d0 & 0xffffu) << 16;                   // tensor_dim0[15:0]
  g1[2] = (tensor_d0 >> 16) | ((tensor_d1 & 0xffffu) << 16);
  g1[3] = (tensor_d1 >> 16) | (tile_d0 << 16);           // tile_dim0
  g1[4] = tile_d1;                                       // tile_dim1, tile_dim2=0
  g1[5] = stride0;                                       // tensor_dim0_stride[31:0]
  g1[6] = 0u;                                            // stride0[47:32], stride1 lo
  g1[7] = 0u;                                            // stride1 hi
  asm volatile("tensor_load_to_lds %0, %1" :: "s"(g0), "s"(g1) : "memory");
}

// ---------------- f32 -> bf16 conversion ----------------
__global__ __launch_bounds__(256) void cvt_f32_bf16(const float* __restrict__ src,
                                                    bf16* __restrict__ dst, size_t n) {
  size_t i = (size_t)blockIdx.x * blockDim.x + threadIdx.x;
  const size_t stride = (size_t)gridDim.x * blockDim.x;
  for (; i < n; i += stride) dst[i] = (bf16)src[i];
}

// ---------------- TDM double-buffered WMMA GEMM ----------------
// C = act(A[MxK] * B[KxN] + bias (+res)).  Block = 256 threads = 8 waves
// (4 in M x 2 in N); per-wave tile 32x64 -> block 128x128.  Wave 0 drives the
// Tensor Data Mover: per 32-deep K-slab it DMAs A(128x32) and B(32x128) into
// LDS (double buffered), overlapping the next slab's DMA with this slab's WMMAs.
template <bool RELU, bool OUT_BF16, bool ADD_RES>
__global__ __launch_bounds__(256) void gemm_bf16_wmma_tdm(
    const bf16* __restrict__ A, const bf16* __restrict__ Bm,
    const float* __restrict__ bias, const float* __restrict__ res,
    void* __restrict__ Cout, int M, int N, int K) {
  __shared__ __align__(128) bf16 Abuf[2][128][32];   // 2 x 8 KB
  __shared__ __align__(128) bf16 Bbuf[2][32][128];   // 2 x 8 KB
  const int lane = threadIdx.x & 31;
  const int wave = threadIdx.x >> 5;
  const int mw = wave >> 1;   // 0..3
  const int nw = wave & 1;    // 0..1
  const size_t mblk = (size_t)blockIdx.y * 128;
  const size_t nblk = (size_t)blockIdx.x * 128;

  v8f acc[2][4] = {};

  if (wave == 0) {  // prologue: slab k=0 into buffer 0
    tdm_load_2d_bf16((uint32_t)(uintptr_t)&Abuf[0][0][0], A + mblk * K,
                     (uint32_t)K, (uint32_t)M, 32u, 128u, (uint32_t)K);
    tdm_load_2d_bf16((uint32_t)(uintptr_t)&Bbuf[0][0][0], Bm + nblk,
                     (uint32_t)N, (uint32_t)K, 128u, 32u, (uint32_t)N);
  }

  int p = 0;
  for (int k = 0; k < K; k += 32, p ^= 1) {
    if (wave == 0) {
      if (k + 32 < K) {  // kick off next slab, then wait only for current pair
        tdm_load_2d_bf16((uint32_t)(uintptr_t)&Abuf[p ^ 1][0][0],
                         A + mblk * K + (k + 32),
                         (uint32_t)K, (uint32_t)M, 32u, 128u, (uint32_t)K);
        tdm_load_2d_bf16((uint32_t)(uintptr_t)&Bbuf[p ^ 1][0][0],
                         Bm + (size_t)(k + 32) * N + nblk,
                         (uint32_t)N, (uint32_t)K, 128u, 32u, (uint32_t)N);
        __builtin_amdgcn_s_wait_tensorcnt(2);
      } else {
        __builtin_amdgcn_s_wait_tensorcnt(0);
      }
    }
    __syncthreads();  // current buffer ready for all 8 waves

    const bf16* As = &Abuf[p][mw * 32][0];   // [32][32], lda = 32
    const bf16* Bs = &Bbuf[p][0][nw * 64];   // [32][64] view, ldb = 128
    v16bf a0 = load_a_frag(As, 32, (size_t)(lane & 15), 0, lane);
    v16bf a1 = load_a_frag(As, 32, (size_t)(16 + (lane & 15)), 0, lane);
#pragma unroll
    for (int j = 0; j < 4; ++j) {
      v16bf bfr = load_b_frag(Bs, 128, 0, j * 16, lane);
      acc[0][j] = wmma_bf16(a0, bfr, acc[0][j]);
      acc[1][j] = wmma_bf16(a1, bfr, acc[1][j]);
    }
    __syncthreads();  // all reads done before this buffer is overwritten
  }

  // C layout: lane -> N = lane%16, vgpr r -> M = r + 8*(lane/16)
  const int ln = lane & 15, lh = lane >> 4;
#pragma unroll
  for (int mt = 0; mt < 2; ++mt)
#pragma unroll
    for (int j = 0; j < 4; ++j)
#pragma unroll
      for (int r = 0; r < 8; ++r) {
        const size_t row = mblk + mw * 32 + mt * 16 + r + 8 * lh;
        const size_t col = nblk + nw * 64 + j * 16 + ln;
        float v = acc[mt][j][r] + bias[col];
        if (ADD_RES) v += res[row * (size_t)N + col];
        if (RELU) v = fmaxf(v, 0.0f);
        if (OUT_BF16) ((bf16*)Cout)[row * (size_t)N + col] = (bf16)v;
        else          ((float*)Cout)[row * (size_t)N + col] = v;
      }
}

// ---------------- per-head QKV projection ----------------
// grid.z = which(0=q,1=k,2=v)*BH + bh ; grid.y tiles T by 256 rows (8 waves x 32)
__global__ __launch_bounds__(256) void qkv_proj_wmma(
    const bf16* __restrict__ xb,
    const bf16* __restrict__ Wqb, const bf16* __restrict__ Wkb,
    const bf16* __restrict__ Wvb,
    const float* __restrict__ bq, const float* __restrict__ bk,
    const float* __restrict__ bv,
    bf16* __restrict__ qo, bf16* __restrict__ kTo, bf16* __restrict__ vo) {
  const int z = blockIdx.z;
  const int which = z / BHc;
  const int bh = z % BHc;
  const int b = bh / Hc, h = bh % Hc;
  const bf16* W; const float* bias;
  if (which == 0)      { W = Wqb; bias = bq; }
  else if (which == 1) { W = Wkb; bias = bk; }
  else                 { W = Wvb; bias = bv; }
  W    += (size_t)h * Dc * NKc;   // [D x 64] row-major, ldb = 64
  bias += (size_t)h * NKc;
  const bf16* xrow = xb + (size_t)b * Tc * Dc;

  const int lane = threadIdx.x & 31;
  const int wave = threadIdx.x >> 5;
  const size_t trow = (size_t)blockIdx.y * 256 + (size_t)wave * 32;
  const size_t r0 = trow + (lane & 15);

  v8f acc[2][4] = {};
  for (int k = 0; k < Dc; k += 32) {
    v16bf a0 = load_a_frag(xrow, Dc, r0,      k, lane);
    v16bf a1 = load_a_frag(xrow, Dc, r0 + 16, k, lane);
#pragma unroll
    for (int j = 0; j < 4; ++j) {
      v16bf bfr = load_b_frag(W, NKc, k, j * 16, lane);
      acc[0][j] = wmma_bf16(a0, bfr, acc[0][j]);
      acc[1][j] = wmma_bf16(a1, bfr, acc[1][j]);
    }
  }
  const int ln = lane & 15, lh = lane >> 4;
  if (which == 1) {  // K stored transposed: kT[bh][64][T]
    bf16* outp = kTo + (size_t)bh * NKc * Tc;
#pragma unroll
    for (int mt = 0; mt < 2; ++mt)
#pragma unroll
      for (int j = 0; j < 4; ++j)
#pragma unroll
        for (int r = 0; r < 8; ++r) {
          const size_t t  = trow + mt * 16 + r + 8 * lh;
          const int    kk = j * 16 + ln;
          outp[(size_t)kk * Tc + t] = (bf16)(acc[mt][j][r] + bias[kk]);
        }
  } else {           // q/v: [bh][T][64]
    bf16* outp = ((which == 0) ? qo : vo) + (size_t)bh * Tc * NKc;
#pragma unroll
    for (int mt = 0; mt < 2; ++mt)
#pragma unroll
      for (int j = 0; j < 4; ++j)
#pragma unroll
        for (int r = 0; r < 8; ++r) {
          const size_t t  = trow + mt * 16 + r + 8 * lh;
          const int    kk = j * 16 + ln;
          outp[t * NKc + kk] = (bf16)(acc[mt][j][r] + bias[kk]);
        }
  }
}

// ---------------- causal flash attention (one wave = 16 query rows) ----------
__global__ __launch_bounds__(256) void attn_flash_wmma(
    const bf16* __restrict__ q, const bf16* __restrict__ kT,
    const bf16* __restrict__ v, bf16* __restrict__ obf) {
  __shared__ __align__(128) bf16 psh[8][16][32];  // per-wave P staging
  const int lane = threadIdx.x & 31;
  const int wave = threadIdx.x >> 5;
  const int bh = blockIdx.y;
  const int b = bh / Hc, h = bh % Hc;
  const int qrow0 = blockIdx.x * 128 + wave * 16;
  const bf16* qp  = q  + (size_t)bh * Tc * NKc;
  const bf16* kTp = kT + (size_t)bh * NKc * Tc;   // [64][T]
  const bf16* vp  = v  + (size_t)bh * Tc * NVc;   // [T][64]
  const int ln = lane & 15, lh = lane >> 4;
  const float scale = 0.125f;  // 1/sqrt(64)

  // Q fragments for this wave's 16 rows (K = 0..31 and 32..63)
  v16bf qa0 = load_a_frag(qp, NKc, (size_t)qrow0 + ln, 0,  lane);
  v16bf qa1 = load_a_frag(qp, NKc, (size_t)qrow0 + ln, 32, lane);

  v8f oacc[4] = {};
  float m_i[8], l_i[8];
#pragma unroll
  for (int r = 0; r < 8; ++r) { m_i[r] = -1e30f; l_i[r] = 0.0f; }

  for (int s0 = 0; s0 < qrow0 + 16; s0 += 32) {
    // scores: two 16x16 tiles covering keys [s0, s0+32)
    v8f sc[2] = {};
#pragma unroll
    for (int c = 0; c < 2; ++c) {
      v16bf bk0 = load_b_frag(kTp, Tc, 0,  s0 + 16 * c, lane);
      sc[c] = wmma_bf16(qa0, bk0, sc[c]);
      v16bf bk1 = load_b_frag(kTp, Tc, 32, s0 + 16 * c, lane);
      sc[c] = wmma_bf16(qa1, bk1, sc[c]);
    }
    // online softmax in C-layout registers; row M is spread across 16 lanes
#pragma unroll
    for (int r = 0; r < 8; ++r) {
      const int t = qrow0 + r + 8 * lh;
      float x0 = sc[0][r] * scale; if (s0 + ln      > t) x0 = -1e30f;
      float x1 = sc[1][r] * scale; if (s0 + 16 + ln > t) x1 = -1e30f;
      float mx = fmaxf(x0, x1);
#pragma unroll
      for (int o = 8; o > 0; o >>= 1) mx = fmaxf(mx, __shfl_xor(mx, o, 16));
      const float mnew = fmaxf(m_i[r], mx);
      const float corr = __expf(m_i[r] - mnew);
      const float p0 = __expf(x0 - mnew);
      const float p1 = __expf(x1 - mnew);
      float sum = p0 + p1;
#pragma unroll
      for (int o = 8; o > 0; o >>= 1) sum += __shfl_xor(sum, o, 16);
      l_i[r] = l_i[r] * corr + sum;
      m_i[r] = mnew;
#pragma unroll
      for (int j = 0; j < 4; ++j) oacc[j][r] *= corr;
      sc[0][r] = p0; sc[1][r] = p1;
    }
    // stage P to LDS (per-wave region; DS ops are in-order within a wave)
#pragma unroll
    for (int r = 0; r < 8; ++r) {
      psh[wave][r + 8 * lh][ln]      = (bf16)sc[0][r];
      psh[wave][r + 8 * lh][16 + ln] = (bf16)sc[1][r];
    }
    // reload P as an A-fragment (16x32)
    const bf16* prow = &psh[wave][ln][0];
    const int hi8 = (lane & 16) ? 8 : 0;
    v8bf plo = *(const v8bf*)(prow + hi8);
    v8bf phi = *(const v8bf*)(prow + 16 + hi8);
    v16bf pa;
#pragma unroll
    for (int i = 0; i < 8; ++i) { pa[i] = plo[i]; pa[8 + i] = phi[i]; }
    // O += P * V
#pragma unroll
    for (int j = 0; j < 4; ++j) {
      v16bf vb = load_b_frag(vp, NVc, s0, j * 16, lane);
      oacc[j] = wmma_bf16(pa, vb, oacc[j]);
    }
  }
  // epilogue: normalize and write concat-head layout o[b][t][h*64 + c] as bf16
#pragma unroll
  for (int j = 0; j < 4; ++j)
#pragma unroll
    for (int r = 0; r < 8; ++r) {
      const size_t t   = (size_t)qrow0 + r + 8 * lh;
      const size_t col = (size_t)h * NVc + j * 16 + ln;
      obf[((size_t)b * Tc + t) * (size_t)(Hc * NVc) + col] =
          (bf16)(oacc[j][r] / l_i[r]);
    }
}

// ---------------- layernorms ----------------
__global__ __launch_bounds__(256) void layernorm_to_bf16(
    const float* __restrict__ in, const float* __restrict__ g,
    const float* __restrict__ bb, bf16* __restrict__ outb) {
  __shared__ float s1[256], s2[256];
  const int row = blockIdx.x;
  const int tid = threadIdx.x;
  const float* xr = in + (size_t)row * Dc;
  float a = 0.f, sq = 0.f;
  for (int i = tid; i < Dc; i += 256) { float t = xr[i]; a += t; sq += t * t; }
  s1[tid] = a; s2[tid] = sq;
  __syncthreads();
  for (int o = 128; o > 0; o >>= 1) {
    if (tid < o) { s1[tid] += s1[tid + o]; s2[tid] += s2[tid + o]; }
    __syncthreads();
  }
  const float mu   = s1[0] * (1.0f / Dc);
  const float var  = s2[0] * (1.0f / Dc) - mu * mu;
  const float rstd = rsqrtf(var + EPSc);
  for (int i = tid; i < Dc; i += 256)
    outb[(size_t)row * Dc + i] = (bf16)((xr[i] - mu) * rstd * g[i] + bb[i]);
}

__global__ __launch_bounds__(256) void layernorm_plus_y(
    const float* __restrict__ yin, const float* __restrict__ g,
    const float* __restrict__ bb, float* __restrict__ outf) {
  __shared__ float s1[256], s2[256];
  const int row = blockIdx.x;
  const int tid = threadIdx.x;
  const float* yr = yin + (size_t)row * Dc;
  float a = 0.f, sq = 0.f;
  for (int i = tid; i < Dc; i += 256) { float t = yr[i]; a += t; sq += t * t; }
  s1[tid] = a; s2[tid] = sq;
  __syncthreads();
  for (int o = 128; o > 0; o >>= 1) {
    if (tid < o) { s1[tid] += s1[tid + o]; s2[tid] += s2[tid + o]; }
    __syncthreads();
  }
  const float mu   = s1[0] * (1.0f / Dc);
  const float var  = s2[0] * (1.0f / Dc) - mu * mu;
  const float rstd = rsqrtf(var + EPSc);
  for (int i = tid; i < Dc; i += 256) {
    const float yv = yr[i];
    outf[(size_t)row * Dc + i] = (yv - mu) * rstd * g[i] + bb[i] + yv;
  }
}

// ---------------- host-side orchestration ----------------
extern "C" void kernel_launch(void* const* d_in, const int* in_sizes, int n_in,
                              void* d_out, int out_size, void* d_ws, size_t ws_size,
                              hipStream_t stream) {
  (void)in_sizes; (void)n_in; (void)out_size; (void)ws_size;
  const float* x   = (const float*)d_in[0];
  const float* Wq  = (const float*)d_in[1];
  const float* bq  = (const float*)d_in[2];
  const float* Wk  = (const float*)d_in[3];
  const float* bk  = (const float*)d_in[4];
  const float* Wv  = (const float*)d_in[5];
  const float* bv  = (const float*)d_in[6];
  const float* Wo  = (const float*)d_in[7];
  const float* bo  = (const float*)d_in[8];
  const float* g1  = (const float*)d_in[9];
  const float* be1 = (const float*)d_in[10];
  const float* W1  = (const float*)d_in[11];
  const float* b1  = (const float*)d_in[12];
  const float* W2  = (const float*)d_in[13];
  const float* b2  = (const float*)d_in[14];
  const float* g2  = (const float*)d_in[15];
  const float* be2 = (const float*)d_in[16];
  float* out = (float*)d_out;

  char* ws = (char*)d_ws;
  size_t off = 0;
  auto take = [&](size_t bytes) -> char* {
    off = (off + 255) & ~(size_t)255;
    char* p = ws + off;
    off += bytes;
    return p;
  };

  const size_t nX  = (size_t)BTc * Dc;
  const size_t nWh = (size_t)Hc * Dc * NKc;     // per-head-stack weight elems
  const size_t nWo = (size_t)(Hc * NVc) * Dc;
  const size_t nW1 = (size_t)Dc * NFc;
  const size_t nW2 = (size_t)NFc * Dc;
  const size_t nQ  = (size_t)BHc * Tc * NKc;

  bf16* xb   = (bf16*)take(nX  * sizeof(bf16));
  bf16* wqb  = (bf16*)take(nWh * sizeof(bf16));
  bf16* wkb  = (bf16*)take(nWh * sizeof(bf16));
  bf16* wvb  = (bf16*)take(nWh * sizeof(bf16));
  bf16* wob  = (bf16*)take(nWo * sizeof(bf16));
  bf16* w1b  = (bf16*)take(nW1 * sizeof(bf16));
  bf16* w2b  = (bf16*)take(nW2 * sizeof(bf16));
  bf16* qb   = (bf16*)take(nQ  * sizeof(bf16));
  bf16* kTb  = (bf16*)take(nQ  * sizeof(bf16));
  bf16* vb   = (bf16*)take(nQ  * sizeof(bf16));
  bf16* ob   = (bf16*)take(nX  * sizeof(bf16));   // concat-head attn out
  float* hpre = (float*)take(nX * sizeof(float)); // x + attn_out
  bf16* hb   = (bf16*)take(nX  * sizeof(bf16));   // LN1 output
  bf16* ub   = (bf16*)take((size_t)BTc * NFc * sizeof(bf16)); // relu(hW1+b1)
  float* yb  = (float*)take(nX * sizeof(float));  // FFN output

  // --- f32 -> bf16 conversions ---
  cvt_f32_bf16<<<2048, 256, 0, stream>>>(x,  xb,  nX);
  cvt_f32_bf16<<<1024, 256, 0, stream>>>(Wq, wqb, nWh);
  cvt_f32_bf16<<<1024, 256, 0, stream>>>(Wk, wkb, nWh);
  cvt_f32_bf16<<<1024, 256, 0, stream>>>(Wv, wvb, nWh);
  cvt_f32_bf16<<<1024, 256, 0, stream>>>(Wo, wob, nWo);
  cvt_f32_bf16<<<2048, 256, 0, stream>>>(W1, w1b, nW1);
  cvt_f32_bf16<<<2048, 256, 0, stream>>>(W2, w2b, nW2);

  // --- QKV projections (K stored transposed) ---
  dim3 gq(1, Tc / 256, 3 * BHc);
  qkv_proj_wmma<<<gq, 256, 0, stream>>>(xb, wqb, wkb, wvb, bq, bk, bv, qb, kTb, vb);

  // --- causal flash attention ---
  dim3 ga(Tc / 128, BHc);
  attn_flash_wmma<<<ga, 256, 0, stream>>>(qb, kTb, vb, ob);

  // --- output projection + bias + residual (f32 out), TDM-pipelined ---
  dim3 go(Dc / 128, BTc / 128);
  gemm_bf16_wmma_tdm<false, false, true><<<go, 256, 0, stream>>>(
      ob, wob, bo, x, hpre, BTc, Dc, Hc * NVc);

  // --- LN1 -> bf16 ---
  layernorm_to_bf16<<<BTc, 256, 0, stream>>>(hpre, g1, be1, hb);

  // --- FFN (TDM-pipelined GEMMs) ---
  dim3 gf1(NFc / 128, BTc / 128);
  gemm_bf16_wmma_tdm<true, true, false><<<gf1, 256, 0, stream>>>(
      hb, w1b, b1, nullptr, ub, BTc, NFc, Dc);
  dim3 gf2(Dc / 128, BTc / 128);
  gemm_bf16_wmma_tdm<false, false, false><<<gf2, 256, 0, stream>>>(
      ub, w2b, b2, nullptr, yb, BTc, Dc, NFc);

  // --- LN2(y) + y -> output ---
  layernorm_plus_y<<<BTc, 256, 0, stream>>>(yb, g2, be2, out);
}
